// GraphDiffNetParallel_46720654246027
// MI455X (gfx1250) — matrix operand order repaired and checked by reading, not compiled, not run on hardware
//
#include <hip/hip_runtime.h>
#include <hip/hip_bf16.h>

// ---------------- problem constants ----------------
#define NV   6000
#define NG   3000
#define KEV  128
#define NE   48000
#define CIN  128
#define CW   64
#define NB   4
#define NVP  6016   // NV padded to multiple of 32
#define NGP  3008   // NG padded to multiple of 32
#define SIGMA 2.5f

typedef __bf16 bh_t;
typedef __attribute__((ext_vector_type(16))) __bf16 v16bf;
typedef __attribute__((ext_vector_type(8)))  float  v8f;
typedef __attribute__((ext_vector_type(4)))  unsigned int v4u;

// =====================================================================
// Big GEMM on the WMMA path:
//   C[M x 64] (f32, row-major, ldc=64) = A[M x K](bf16 row-major, lda=K)
//                                        @ B[K x 64], B given as Bt[64 x K]
// Requirements: M % 32 == 0, K % 32 == 0, 16B-aligned rows.
// Each wave computes a 32-row x 64-col strip: 8 accumulators, so per
// K-step we do 2 A-fragment loads + 4 B-fragment loads for 8 WMMAs
// (A bytes fetched exactly once per GEMM -> memory-bound optimum).
// Block = 128 threads = 4 waves = 128 rows.
// =====================================================================
__global__ void __launch_bounds__(128)
wmma_gemm_n64(const bh_t* __restrict__ A, const bh_t* __restrict__ Bt,
              float* __restrict__ C, int M, int K) {
    const int wave = threadIdx.x >> 5;
    const int lane = threadIdx.x & 31;
    const int half = lane >> 4;          // 0: lanes 0-15, 1: lanes 16-31
    const int l16  = lane & 15;
    const int m0   = (blockIdx.x * 4 + wave) * 32;   // 32 rows per wave
    if (m0 >= M) return;

    const bh_t* arow0 = A + (size_t)(m0 + l16) * K;        // A rows m0..m0+15
    const bh_t* arow1 = A + (size_t)(m0 + 16 + l16) * K;   // A rows m0+16..m0+31
    const bh_t* bc0 = Bt + (size_t)(l16) * K;              // B columns for N-tile j
    const bh_t* bc1 = Bt + (size_t)(16 + l16) * K;
    const bh_t* bc2 = Bt + (size_t)(32 + l16) * K;
    const bh_t* bc3 = Bt + (size_t)(48 + l16) * K;

    v8f acc0 = {}, acc1 = {}, acc2 = {}, acc3 = {};
    v8f acc4 = {}, acc5 = {}, acc6 = {}, acc7 = {};

    for (int k0 = 0; k0 < K; k0 += 32) {
        union { v16bf v; v4u q[2]; } a0, a1, b0, b1, b2, b3;
        // A 16x32 fragment, 16-bit layout: half0 -> K {0..7,16..23}, half1 -> K {8..15,24..31}
        const int ao = k0 + 8 * half;
        a0.q[0] = *(const v4u*)(arow0 + ao);
        a0.q[1] = *(const v4u*)(arow0 + ao + 16);
        a1.q[0] = *(const v4u*)(arow1 + ao);
        a1.q[1] = *(const v4u*)(arow1 + ao + 16);
        // B 32x16 fragments, same layout as A: half0 -> K {0..7,16..23}, half1 -> K {8..15,24..31}
        const int bo = k0 + 8 * half;
        b0.q[0] = *(const v4u*)(bc0 + bo);  b0.q[1] = *(const v4u*)(bc0 + bo + 16);
        b1.q[0] = *(const v4u*)(bc1 + bo);  b1.q[1] = *(const v4u*)(bc1 + bo + 16);
        b2.q[0] = *(const v4u*)(bc2 + bo);  b2.q[1] = *(const v4u*)(bc2 + bo + 16);
        b3.q[0] = *(const v4u*)(bc3 + bo);  b3.q[1] = *(const v4u*)(bc3 + bo + 16);

        acc0 = __builtin_amdgcn_wmma_f32_16x16x32_bf16(false, a0.v, false, b0.v, (short)0, acc0, false, false);
        acc1 = __builtin_amdgcn_wmma_f32_16x16x32_bf16(false, a0.v, false, b1.v, (short)0, acc1, false, false);
        acc2 = __builtin_amdgcn_wmma_f32_16x16x32_bf16(false, a0.v, false, b2.v, (short)0, acc2, false, false);
        acc3 = __builtin_amdgcn_wmma_f32_16x16x32_bf16(false, a0.v, false, b3.v, (short)0, acc3, false, false);
        acc4 = __builtin_amdgcn_wmma_f32_16x16x32_bf16(false, a1.v, false, b0.v, (short)0, acc4, false, false);
        acc5 = __builtin_amdgcn_wmma_f32_16x16x32_bf16(false, a1.v, false, b1.v, (short)0, acc5, false, false);
        acc6 = __builtin_amdgcn_wmma_f32_16x16x32_bf16(false, a1.v, false, b2.v, (short)0, acc6, false, false);
        acc7 = __builtin_amdgcn_wmma_f32_16x16x32_bf16(false, a1.v, false, b3.v, (short)0, acc7, false, false);
    }

    // C layout: VGPR r of a tile -> row tile_m0 + 8*half + r, col tile_n0 + l16
    float* c0 = C + (size_t)(m0 + 8 * half) * 64 + l16;        // row strip 0
    float* c1 = C + (size_t)(m0 + 16 + 8 * half) * 64 + l16;   // row strip 1
#pragma unroll
    for (int r = 0; r < 8; ++r) {
        size_t ro = (size_t)r * 64;
        c0[ro]      = acc0[r];
        c0[ro + 16] = acc1[r];
        c0[ro + 32] = acc2[r];
        c0[ro + 48] = acc3[r];
        c1[ro]      = acc4[r];
        c1[ro + 16] = acc5[r];
        c1[ro + 32] = acc6[r];
        c1[ro + 48] = acc7[r];
    }
}

// =====================================================================
// Small / elementwise kernels (fp32)
// =====================================================================
__global__ void k_fill(float* p, float v, int n) {
    int i = blockIdx.x * blockDim.x + threadIdx.x;
    if (i < n) p[i] = v;
}

// Y[m,n] = act( X[m,:] . W[n,:] + b[n] (+ skip[m,n]) )
__global__ void k_linear(const float* __restrict__ X, int ldx,
                         const float* __restrict__ W, int kdim,
                         const float* __restrict__ b,
                         const float* __restrict__ skip,
                         float* __restrict__ Y,
                         int M, int N, int relu) {
    int idx = blockIdx.x * blockDim.x + threadIdx.x;
    if (idx >= M * N) return;
    int m = idx / N, n = idx % N;
    const float* x = X + (size_t)m * ldx;
    const float* w = W + (size_t)n * kdim;
    float acc = b ? b[n] : 0.f;
    for (int k = 0; k < kdim; ++k) acc = fmaf(x[k], w[k], acc);
    if (skip) acc += skip[(size_t)m * N + n];
    if (relu) acc = fmaxf(acc, 0.f);
    Y[(size_t)m * N + n] = acc;
}

// xspec[k,c] = sum_v evecs[v,k] * mass[v] * x[v,c]
__global__ void k_tobasis(const float* __restrict__ evecs, const float* __restrict__ x,
                          const float* __restrict__ mass, float* __restrict__ xspec) {
    int idx = blockIdx.x * blockDim.x + threadIdx.x;
    if (idx >= KEV * CW) return;
    int k = idx / CW, c = idx % CW;
    float acc = 0.f;
    for (int v = 0; v < NV; ++v)
        acc = fmaf(evecs[(size_t)v * KEV + k] * mass[v], x[(size_t)v * CW + c], acc);
    xspec[(size_t)k * CW + c] = acc;
}

// xdiff[v,c] = sum_k evecs[v,k] * exp(-evals[k]*max(t[c],1e-8)) * xspec[k,c]
__global__ void k_frombasis(const float* __restrict__ evecs, const float* __restrict__ xspec,
                            const float* __restrict__ evals, const float* __restrict__ tvec,
                            float* __restrict__ xdiff) {
    int idx = blockIdx.x * blockDim.x + threadIdx.x;
    if (idx >= NV * CW) return;
    int v = idx / CW, c = idx % CW;
    float t = fmaxf(tvec[c], 1e-8f);
    float acc = 0.f;
    for (int k = 0; k < KEV; ++k)
        acc = fmaf(evecs[(size_t)v * KEV + k] * __expf(-evals[k] * t),
                   xspec[(size_t)k * CW + c], acc);
    xdiff[(size_t)v * CW + c] = acc;
}

// out[c*Mp + m] = (m<M) ? (bf16) in[m*64 + c] : 0   (B-operand pre-transpose + pad)
__global__ void k_transpose_cvt(const float* __restrict__ in, int M, int Mp,
                                bh_t* __restrict__ out) {
    int idx = blockIdx.x * blockDim.x + threadIdx.x;
    if (idx >= CW * Mp) return;
    int c = idx / Mp, m = idx % Mp;
    out[(size_t)c * Mp + m] = (m < M) ? (bh_t)in[(size_t)m * CW + c] : (bh_t)0.f;
}

// out[Mp x Np] (bf16, zero-padded) = in[M x N] (f32)
__global__ void k_cvt_pad(const float* __restrict__ in, int M, int N, int Np,
                          bh_t* __restrict__ out, long long total) {
    long long idx = (long long)blockIdx.x * blockDim.x + threadIdx.x;
    if (idx >= total) return;
    int i = (int)(idx / Np), j = (int)(idx % Np);
    out[idx] = (i < M && j < N) ? (bh_t)in[(size_t)i * N + j] : (bh_t)0.f;
}

// rbf[i,j] = exp(-|v_i - p_j| / SIGMA), stored bf16 both orders, zero pad
__global__ void k_rbf(const float* __restrict__ verts, const float* __restrict__ pos,
                      bh_t* __restrict__ rbfb, bh_t* __restrict__ rbfTb) {
    long long idx = (long long)blockIdx.x * blockDim.x + threadIdx.x;
    if (idx >= (long long)NVP * NGP) return;
    int i = (int)(idx / NGP), j = (int)(idx % NGP);
    float w = 0.f;
    if (i < NV && j < NG) {
        float dx = verts[i * 3 + 0] - pos[j * 3 + 0];
        float dy = verts[i * 3 + 1] - pos[j * 3 + 1];
        float dz = verts[i * 3 + 2] - pos[j * 3 + 2];
        float dist = sqrtf(fmaxf(dx * dx + dy * dy + dz * dz, 0.f));
        w = __expf(-dist / SIGMA);
    }
    rbfb[(size_t)i * NGP + j]  = (bh_t)w;
    rbfTb[(size_t)j * NVP + i] = (bh_t)w;
}

// gfeat = tanh(gX*br + gY*bi),  br = gX.A_re^T - gY.A_im^T,  bi = gY.A_re^T + gX.A_im^T
__global__ void k_gfeat(const float* __restrict__ gX, const float* __restrict__ gY,
                        const float* __restrict__ Are, const float* __restrict__ Aim,
                        float* __restrict__ gfeat) {
    int idx = blockIdx.x * blockDim.x + threadIdx.x;
    if (idx >= NV * CW) return;
    int m = idx / CW, n = idx % CW;
    const float* ar = Are + (size_t)n * CW;
    const float* ai = Aim + (size_t)n * CW;
    const float* gx = gX + (size_t)m * CW;
    const float* gy = gY + (size_t)m * CW;
    float br = 0.f, bi = 0.f;
    for (int k = 0; k < CW; ++k) {
        br = fmaf(gx[k], ar[k], br) - gy[k] * ai[k];
        bi = fmaf(gy[k], ar[k], bi) + gx[k] * ai[k];
    }
    gfeat[(size_t)m * CW + n] = tanhf(gx[n] * br + gy[n] * bi);
}

// h = concat([dx, xdiff, gfeat], -1)  -> (NV, 3*CW)
__global__ void k_build_h(const float* __restrict__ dx, const float* __restrict__ xdiff,
                          const float* __restrict__ gfeat, float* __restrict__ h) {
    int idx = blockIdx.x * blockDim.x + threadIdx.x;
    if (idx >= NV * 3 * CW) return;
    int m = idx / (3 * CW), c = idx % (3 * CW);
    float v;
    if (c < CW)            v = dx[(size_t)m * CW + c];
    else if (c < 2 * CW)   v = xdiff[(size_t)m * CW + (c - CW)];
    else                   v = gfeat[(size_t)m * CW + (c - 2 * CW)];
    h[(size_t)m * 3 * CW + c] = v;
}

// ---- GCN helpers ----
__global__ void k_deg(const int* __restrict__ dst, float* __restrict__ deg) {
    int e = blockIdx.x * blockDim.x + threadIdx.x;
    if (e < NE) atomicAdd(&deg[dst[e]], 1.0f);
}
__global__ void k_dinv(const float* __restrict__ deg, float* __restrict__ dinv) {
    int i = blockIdx.x * blockDim.x + threadIdx.x;
    if (i < NG) dinv[i] = rsqrtf(deg[i]);
}
__global__ void k_gcn_scatter(const int* __restrict__ src, const int* __restrict__ dst,
                              const float* __restrict__ dinv, const float* __restrict__ xw,
                              float* __restrict__ agg) {
    int e = blockIdx.x * blockDim.x + threadIdx.x;
    if (e >= NE) return;
    int s = src[e], d = dst[e];
    float nrm = dinv[s] * dinv[d];
    const float* xs = xw + (size_t)s * CW;
    float* ad = agg + (size_t)d * CW;
    for (int c = 0; c < CW; ++c) atomicAdd(&ad[c], nrm * xs[c]);
}
__global__ void k_gcn_combine(const float* __restrict__ agg, const float* __restrict__ xw,
                              const float* __restrict__ dinv, const float* __restrict__ b,
                              float* __restrict__ out, int relu) {
    int idx = blockIdx.x * blockDim.x + threadIdx.x;
    if (idx >= NG * CW) return;
    int m = idx / CW, n = idx % CW;
    float di = dinv[m];
    float v = agg[idx] + di * di * xw[idx] + b[n];
    out[idx] = relu ? fmaxf(v, 0.f) : v;
}

// mix: out[m,n] = b[n] + Xa[m,:].W[n,0:64] + Xb[m,:].W[n,64:128]
__global__ void k_mix(const float* __restrict__ Xa, const float* __restrict__ Xb,
                      const float* __restrict__ W, const float* __restrict__ b,
                      float* __restrict__ out, int M) {
    int idx = blockIdx.x * blockDim.x + threadIdx.x;
    if (idx >= M * CW) return;
    int m = idx / CW, n = idx % CW;
    const float* w = W + (size_t)n * (2 * CW);
    const float* xa = Xa + (size_t)m * CW;
    const float* xb = Xb + (size_t)m * CW;
    float acc = b[n];
    for (int c = 0; c < CW; ++c) acc = fmaf(xa[c], w[c], acc);
    for (int c = 0; c < CW; ++c) acc = fmaf(xb[c], w[CW + c], acc);
    out[(size_t)m * CW + n] = acc;
}

// =====================================================================
// Host-side orchestration
// =====================================================================
static inline int cdiv(long long a, long long b) { return (int)((a + b - 1) / b); }

extern "C" void kernel_launch(void* const* d_in, const int* in_sizes, int n_in,
                              void* d_out, int out_size, void* d_ws, size_t ws_size,
                              hipStream_t stream) {
    // ---- inputs (setup_inputs order) ----
    const float* surf_x  = (const float*)d_in[0];
    const float* graph_x = (const float*)d_in[1];
    const float* verts   = (const float*)d_in[2];
    const float* pos     = (const float*)d_in[3];
    const float* mass    = (const float*)d_in[4];
    const float* evals   = (const float*)d_in[5];
    const float* evecs   = (const float*)d_in[6];
    const float* gradX   = (const float*)d_in[7];
    const float* gradY   = (const float*)d_in[8];
    const float* lin1_W  = (const float*)d_in[9];
    const float* lin1_b  = (const float*)d_in[10];
    const float* lin2_W  = (const float*)d_in[11];
    const float* lin2_b  = (const float*)d_in[12];
    const float* dtime   = (const float*)d_in[13];
    const float* A_re    = (const float*)d_in[14];
    const float* A_im    = (const float*)d_in[15];
    const float* mlpW0   = (const float*)d_in[16];
    const float* mlpb0   = (const float*)d_in[17];
    const float* mlpW1   = (const float*)d_in[18];
    const float* mlpb1   = (const float*)d_in[19];
    const float* mlpW2   = (const float*)d_in[20];
    const float* mlpb2   = (const float*)d_in[21];
    const float* gcnW1   = (const float*)d_in[22];
    const float* gcnb1   = (const float*)d_in[23];
    const float* gcnW2   = (const float*)d_in[24];
    const float* gcnb2   = (const float*)d_in[25];
    const float* mixW    = (const float*)d_in[26];
    const float* mixb    = (const float*)d_in[27];
    const int*   eidx    = (const int*)d_in[28];
    const int* esrc = eidx;
    const int* edst = eidx + NE;
    float* out = (float*)d_out;

    // ---- workspace bump allocator ----
    char* wp = (char*)d_ws;
    size_t off = 0;
    auto alloc = [&](size_t bytes) -> void* {
        void* p = wp + off;
        off += (bytes + 255) & ~(size_t)255;
        return p;
    };
    bh_t* gradXb = (bh_t*)alloc((size_t)NVP * NVP * 2);
    bh_t* gradYb = (bh_t*)alloc((size_t)NVP * NVP * 2);
    bh_t* rbfb   = (bh_t*)alloc((size_t)NVP * NGP * 2);
    bh_t* rbfTb  = (bh_t*)alloc((size_t)NGP * NVP * 2);
    bh_t* xdT_b  = (bh_t*)alloc((size_t)CW * NVP * 2);
    bh_t* dxT_b  = (bh_t*)alloc((size_t)CW * NVP * 2);
    bh_t* gxT_b  = (bh_t*)alloc((size_t)CW * NGP * 2);
    float* dxA   = (float*)alloc((size_t)NV * CW * 4);
    float* dxB   = (float*)alloc((size_t)NV * CW * 4);
    float* gxA   = (float*)alloc((size_t)NG * CW * 4);
    float* gxB   = (float*)alloc((size_t)NG * CW * 4);
    float* xspec = (float*)alloc((size_t)KEV * CW * 4);
    float* xdiff = (float*)alloc((size_t)NV * CW * 4);
    float* gXb   = (float*)alloc((size_t)NVP * CW * 4);
    float* gYb   = (float*)alloc((size_t)NVP * CW * 4);
    float* gfeat = (float*)alloc((size_t)NV * CW * 4);
    float* hbuf  = (float*)alloc((size_t)NV * 3 * CW * 4);
    float* t1    = (float*)alloc((size_t)NV * CW * 4);
    float* t2    = (float*)alloc((size_t)NV * CW * 4);
    float* dxd   = (float*)alloc((size_t)NV * CW * 4);
    float* xw    = (float*)alloc((size_t)NG * CW * 4);
    float* agg   = (float*)alloc((size_t)NG * CW * 4);
    float* hg    = (float*)alloc((size_t)NG * CW * 4);
    float* gxn   = (float*)alloc((size_t)NG * CW * 4);
    float* deg   = (float*)alloc((size_t)NG * 4);
    float* dinv  = (float*)alloc((size_t)NG * 4);
    float* dog   = (float*)alloc((size_t)NGP * CW * 4);  // diff_on_graph
    float* god   = (float*)alloc((size_t)NVP * CW * 4);  // graph_on_diff
    (void)ws_size; (void)n_in; (void)in_sizes; (void)out_size;

    const int T = 256;

    // ---- one-time prep: bf16 copies, rbf, initial branches, degrees ----
    {
        long long tot = (long long)NVP * NVP;
        k_cvt_pad<<<cdiv(tot, T), T, 0, stream>>>(gradX, NV, NV, NVP, gradXb, tot);
        k_cvt_pad<<<cdiv(tot, T), T, 0, stream>>>(gradY, NV, NV, NVP, gradYb, tot);
        long long rtot = (long long)NVP * NGP;
        k_rbf<<<cdiv(rtot, T), T, 0, stream>>>(verts, pos, rbfb, rbfTb);
    }
    k_linear<<<cdiv((long long)NV * CW, T), T, 0, stream>>>(surf_x, 5, lin1_W, 5, lin1_b,
                                                           nullptr, dxA, NV, CW, 0);
    k_linear<<<cdiv((long long)NG * CW, T), T, 0, stream>>>(graph_x, CIN, lin2_W, CIN, lin2_b,
                                                            nullptr, gxA, NG, CW, 0);
    k_fill<<<cdiv(NG, T), T, 0, stream>>>(deg, 1.0f, NG);
    k_deg<<<cdiv(NE, T), T, 0, stream>>>(edst, deg);
    k_dinv<<<cdiv(NG, T), T, 0, stream>>>(deg, dinv);

    float* dx = dxA; float* dxn = dxB;
    float* gx = gxA; float* gxnext = gxB;

    for (int i = 0; i < NB; ++i) {
        const float* tvec = dtime + (size_t)i * CW;
        const float* Are  = A_re + (size_t)i * CW * CW;
        const float* Aim  = A_im + (size_t)i * CW * CW;
        const float* W0 = mlpW0 + (size_t)i * CW * 3 * CW;
        const float* b0 = mlpb0 + (size_t)i * CW;
        const float* W1 = mlpW1 + (size_t)i * CW * CW;
        const float* b1 = mlpb1 + (size_t)i * CW;
        const float* W2 = mlpW2 + (size_t)i * CW * CW;
        const float* b2 = mlpb2 + (size_t)i * CW;
        const float* gW1 = gcnW1 + (size_t)i * CW * CW;
        const float* gb1 = gcnb1 + (size_t)i * CW;
        const float* gW2 = gcnW2 + (size_t)i * CW * CW;
        const float* gb2 = gcnb2 + (size_t)i * CW;
        const float* mW = mixW + (size_t)i * CW * 2 * CW;
        const float* mb = mixb + (size_t)i * CW;

        // ---- diffusion block ----
        k_tobasis<<<cdiv(KEV * CW, T), T, 0, stream>>>(evecs, dx, mass, xspec);
        k_frombasis<<<cdiv(NV * CW, T), T, 0, stream>>>(evecs, xspec, evals, tvec, xdiff);
        k_transpose_cvt<<<cdiv(CW * NVP, T), T, 0, stream>>>(xdiff, NV, NVP, xdT_b);
        // gX = gradX @ xdiff ; gY = gradY @ xdiff    (WMMA bf16, f32 accumulate)
        wmma_gemm_n64<<<cdiv(NVP, 128), 128, 0, stream>>>(gradXb, xdT_b, gXb, NVP, NVP);
        wmma_gemm_n64<<<cdiv(NVP, 128), 128, 0, stream>>>(gradYb, xdT_b, gYb, NVP, NVP);
        k_gfeat<<<cdiv(NV * CW, T), T, 0, stream>>>(gXb, gYb, Are, Aim, gfeat);
        k_build_h<<<cdiv(NV * 3 * CW, T), T, 0, stream>>>(dx, xdiff, gfeat, hbuf);
        k_linear<<<cdiv(NV * CW, T), T, 0, stream>>>(hbuf, 3 * CW, W0, 3 * CW, b0, nullptr, t1, NV, CW, 1);
        k_linear<<<cdiv(NV * CW, T), T, 0, stream>>>(t1, CW, W1, CW, b1, nullptr, t2, NV, CW, 1);
        k_linear<<<cdiv(NV * CW, T), T, 0, stream>>>(t2, CW, W2, CW, b2, dx, dxd, NV, CW, 0); // + skip

        // ---- GCN conv 1 (relu) ----
        k_linear<<<cdiv(NG * CW, T), T, 0, stream>>>(gx, CW, gW1, CW, nullptr, nullptr, xw, NG, CW, 0);
        k_fill<<<cdiv(NG * CW, T), T, 0, stream>>>(agg, 0.f, NG * CW);
        k_gcn_scatter<<<cdiv(NE, T), T, 0, stream>>>(esrc, edst, dinv, xw, agg);
        k_gcn_combine<<<cdiv(NG * CW, T), T, 0, stream>>>(agg, xw, dinv, gb1, hg, 1);
        // ---- GCN conv 2 ----
        k_linear<<<cdiv(NG * CW, T), T, 0, stream>>>(hg, CW, gW2, CW, nullptr, nullptr, xw, NG, CW, 0);
        k_fill<<<cdiv(NG * CW, T), T, 0, stream>>>(agg, 0.f, NG * CW);
        k_gcn_scatter<<<cdiv(NE, T), T, 0, stream>>>(esrc, edst, dinv, xw, agg);
        k_gcn_combine<<<cdiv(NG * CW, T), T, 0, stream>>>(agg, xw, dinv, gb2, gxn, 0);

        // ---- cross-domain rbf products (WMMA) ----
        k_transpose_cvt<<<cdiv(CW * NVP, T), T, 0, stream>>>(dxd, NV, NVP, dxT_b);
        k_transpose_cvt<<<cdiv(CW * NGP, T), T, 0, stream>>>(gxn, NG, NGP, gxT_b);
        // diff_on_graph = rbf^T @ dxd   : A = rbfT (NGP x NVP)
        wmma_gemm_n64<<<cdiv(NGP, 128), 128, 0, stream>>>(rbfTb, dxT_b, dog, NGP, NVP);
        // graph_on_diff = rbf @ gxn     : A = rbf (NVP x NGP)
        wmma_gemm_n64<<<cdiv(NVP, 128), 128, 0, stream>>>(rbfb, gxT_b, god, NVP, NGP);

        // ---- mix ----
        float* dx_dst = (i == NB - 1) ? out : dxn;
        k_mix<<<cdiv(NV * CW, T), T, 0, stream>>>(dxd, god, mW, mb, dx_dst, NV);
        k_mix<<<cdiv(NG * CW, T), T, 0, stream>>>(dog, gxn, mW, mb, gxnext, NG);

        // ping-pong
        float* tswap;
        tswap = dx; dx = dxn; dxn = tswap;
        tswap = gx; gx = gxnext; gxnext = tswap;
    }
}
